// IndependentLSTMText_7490422964286
// MI455X (gfx1250) — matrix operand order repaired and checked by reading, not compiled, not run on hardware
//
#include <hip/hip_runtime.h>
#include <hip/hip_bf16.h>

typedef __bf16 bf16;
typedef __attribute__((ext_vector_type(16))) __bf16 v16bf;
typedef __attribute__((ext_vector_type(8)))  __bf16 bf16x8;
typedef __attribute__((ext_vector_type(8)))  float  v8f;
typedef __attribute__((ext_vector_type(4)))  int    v4i;

typedef __attribute__((address_space(1))) v4i gv4i;
typedef __attribute__((address_space(3))) v4i lv4i;

#define V_   49408
#define B_   16
#define N_   12
#define L_   12
#define E_   512
#define H_   512
#define G4H  2048
#define M1   192   // B*L rows per LSTM
#define LDST 40    // padded LDS row stride (bf16 elems): 80B, breaks bank patterns

union FragU { v16bf v; bf16x8 h[2]; };

// A-matrix 16x32 bf16 fragment from row-major memory (lane&15 = M row;
// lane>>4 selects K-chunk interleave: K in {hf*8..+7, 16+hf*8..+7}).
__device__ __forceinline__ v16bf load_fragA(const bf16* p, int ld) {
  int lane = threadIdx.x & 31;
  int r = lane & 15, hf = lane >> 4;
  const bf16* q = p + r * ld + hf * 8;
  FragU u;
  u.h[0] = *(const bf16x8*)(q);
  u.h[1] = *(const bf16x8*)(q + 16);
  return u.v;
}

// B-matrix 32x16 bf16 fragment. Column c = lane&15 = row of W (computing
// X @ W^T, W row-major, K contiguous). lane>>4 selects K-chunk interleave: K in {hf*8..+7, 16+hf*8..+7}.
__device__ __forceinline__ v16bf load_fragB(const bf16* p, int ld) {
  int lane = threadIdx.x & 31;
  int c = lane & 15, hf = lane >> 4;
  const bf16* q = p + c * ld + hf * 8;
  FragU u;
  u.h[0] = *(const bf16x8*)(q);
  u.h[1] = *(const bf16x8*)(q + 16);
  return u.v;
}

__device__ __forceinline__ v8f wmma_bf16(v16bf a, v16bf b, v8f c) {
  return __builtin_amdgcn_wmma_f32_16x16x32_bf16(false, a, false, b, (short)0, c, false, false);
}

__device__ __forceinline__ float sigf(float x) { return 1.0f / (1.0f + __expf(-x)); }

// ---- CDNA5 async global->LDS copy (16B per lane), with portable fallback ----
// Builtin signature (from toolchain diagnostic): (v4i AS1*, v4i AS3*, Imm, Imm)
__device__ __forceinline__ void async_copy16(const bf16* g, bf16* l) {
#if __has_builtin(__builtin_amdgcn_global_load_async_to_lds_b128)
  __builtin_amdgcn_global_load_async_to_lds_b128(
      (gv4i*)(v4i*)(void*)g, (lv4i*)(v4i*)(void*)l, 0, 0);
#else
  *(bf16x8*)l = *(const bf16x8*)g;
#endif
}
__device__ __forceinline__ void async_wait_le4() {
#if __has_builtin(__builtin_amdgcn_s_wait_asynccnt)
  __builtin_amdgcn_s_wait_asynccnt(4);
#endif
}
__device__ __forceinline__ void async_wait_le0() {
#if __has_builtin(__builtin_amdgcn_s_wait_asynccnt)
  __builtin_amdgcn_s_wait_asynccnt(0);
#endif
}

// ---------------- prep kernels ----------------

__global__ void k_f32_to_bf16(const float* __restrict__ src, bf16* __restrict__ dst, int n) {
  int i = blockIdx.x * blockDim.x + threadIdx.x;
  int stride = gridDim.x * blockDim.x;
  for (; i < n; i += stride) dst[i] = (bf16)src[i];
}

// Build X[(n*192 + b*12 + t), e]: t==0 -> features[b,n,e]; else emb[n, ids[b,n,t-1], e]
__global__ void k_gather(const float* __restrict__ features, const float* __restrict__ emb,
                         const int* __restrict__ ids, bf16* __restrict__ X) {
  int idx = blockIdx.x * blockDim.x + threadIdx.x;   // exactly N*192*512 threads
  int e = idx & (E_ - 1);
  int m = (idx >> 9) % M1;
  int n = idx / (M1 * E_);
  int b = m / L_, t = m % L_;
  float v;
  if (t == 0) v = features[(b * N_ + n) * E_ + e];
  else        v = emb[((size_t)n * V_ + ids[(b * N_ + n) * L_ + (t - 1)]) * H_ + e];
  X[idx] = (bf16)v;
}

__global__ void k_prep(const float* __restrict__ bih, const float* __restrict__ bhh,
                       float* __restrict__ bb, const unsigned char* __restrict__ label_mask,
                       int* __restrict__ lengths) {
  int i = blockIdx.x * blockDim.x + threadIdx.x;
  if (i < N_ * G4H) bb[i] = bih[i] + bhh[i];
  if (i < B_ * N_) {
    int s = 0;
    for (int l = 0; l < L_; l++) s += label_mask[i * L_ + l] ? 1 : 0;
    lengths[i] = s;
  }
}

// ---------------- GEMM1: xg = X @ Wih^T + (bih+bhh) ----------------
// one 16x16 tile per wave; tiles = 12 n * 12 mt * 128 ct = 18432

__global__ void __launch_bounds__(256) k_gemm1(const bf16* __restrict__ X,
                                               const bf16* __restrict__ W,
                                               const float* __restrict__ bb,
                                               float* __restrict__ xg) {
  int wave = (blockIdx.x * blockDim.x + threadIdx.x) >> 5;
  int ct = wave & 127;
  int mt = (wave >> 7) % 12;
  int n  = wave / (12 * 128);
  const bf16* A  = X + (size_t)(n * M1 + mt * 16) * E_;
  const bf16* Bm = W + ((size_t)n * G4H + ct * 16) * E_;
  v8f acc = {};
#pragma unroll
  for (int k = 0; k < E_; k += 32)
    acc = wmma_bf16(load_fragA(A + k, E_), load_fragB(Bm + k, E_), acc);
  int lane = threadIdx.x & 31;
  int col = lane & 15, hf = lane >> 4;
  float bias = bb[n * G4H + ct * 16 + col];
  float* out = xg + (size_t)(n * M1 + mt * 16) * G4H + ct * 16 + col;
#pragma unroll
  for (int i = 0; i < 8; i++) out[(size_t)(i + 8 * hf) * G4H] = acc[i] + bias;
}

// ---------------- LSTM recurrence ----------------
// block = one LSTM (n). 32 waves; wave w owns hidden cols [16w,16w+16) across
// all 4 gates, so cell state c lives in registers for the whole time loop.

__global__ void __launch_bounds__(1024) k_lstm(const float* __restrict__ h0,
                                               const float* __restrict__ c0,
                                               const bf16* __restrict__ Whh,
                                               const float* __restrict__ xg,
                                               const int* __restrict__ lengths,
                                               const unsigned char* __restrict__ tmask,
                                               bf16* __restrict__ outs) {
  __shared__ __align__(16) bf16 hlds[B_ * H_];     // 16 KB
  int n = blockIdx.x;
  int tid = threadIdx.x;
  int wave = tid >> 5;                             // 0..31 -> hidden-col tile
  int lane = tid & 31;
  int col = lane & 15, hf = lane >> 4;
  int hh = wave * 16 + col;

  for (int i = tid; i < B_ * H_; i += blockDim.x) hlds[i] = (bf16)h0[i];

  float creg[8];
  int len8[8], msk8[8];
#pragma unroll
  for (int i = 0; i < 8; i++) {
    int b = i + 8 * hf;
    creg[i] = c0[b * H_ + hh];
    len8[i] = lengths[b * N_ + n];
    msk8[i] = tmask[b * N_ + n] ? 1 : 0;
  }

  const bf16* Wn = Whh + (size_t)n * G4H * H_;
  const bf16* Bi = Wn + (size_t)(0 * H_ + wave * 16) * H_;
  const bf16* Bf = Wn + (size_t)(1 * H_ + wave * 16) * H_;
  const bf16* Bg = Wn + (size_t)(2 * H_ + wave * 16) * H_;
  const bf16* Bo = Wn + (size_t)(3 * H_ + wave * 16) * H_;

  __syncthreads();
  for (int t = 0; t < L_; t++) {
    v8f ai = {}, af = {}, ag = {}, ao = {};
#pragma unroll
    for (int k = 0; k < H_; k += 32) {
      v16bf A = load_fragA(hlds + k, H_);
      ai = wmma_bf16(A, load_fragB(Bi + k, H_), ai);
      af = wmma_bf16(A, load_fragB(Bf + k, H_), af);
      ag = wmma_bf16(A, load_fragB(Bg + k, H_), ag);
      ao = wmma_bf16(A, load_fragB(Bo + k, H_), ao);
    }
    float hval[8];
#pragma unroll
    for (int i = 0; i < 8; i++) {
      int b = i + 8 * hf;
      const float* xr = xg + (size_t)(n * M1 + b * L_ + t) * G4H;
      float gi = sigf(ai[i] + xr[0 * H_ + hh]);
      float gf = sigf(af[i] + xr[1 * H_ + hh]);
      float gg = tanhf(ag[i] + xr[2 * H_ + hh]);
      float go = sigf(ao[i] + xr[3 * H_ + hh]);
      float c = gf * creg[i] + gi * gg;
      creg[i] = c;
      hval[i] = go * tanhf(c);
    }
    __syncthreads();   // all waves done reading hlds for this step
#pragma unroll
    for (int i = 0; i < 8; i++) {
      int b = i + 8 * hf;
      hlds[b * H_ + hh] = (bf16)hval[i];
      float hv = (msk8[i] && (t < len8[i])) ? hval[i] : 0.0f;
      outs[((size_t)(b * N_ + n) * L_ + t) * H_ + hh] = (bf16)hv;
    }
    __syncthreads();   // new h visible before next step
  }
}

// ---------------- GEMM2: logits = outs @ Wout^T + bout ----------------
// M=2304, N=49408, K=512. Block tile 128x128, 8 waves (4 row x 2 col),
// wave tile 32x64. K staged 32 at a time through double-buffered LDS using
// async global->LDS copies (ASYNCcnt), 16 stages.

__global__ void __launch_bounds__(256) k_gemm2(const bf16* __restrict__ Aouts,
                                               const bf16* __restrict__ Wb,
                                               const float* __restrict__ bout,
                                               float* __restrict__ out) {
  __shared__ __align__(16) bf16 Asl[2][128 * LDST];   // 2 x 10KB
  __shared__ __align__(16) bf16 Bsl[2][128 * LDST];   // 2 x 10KB

  int tid  = threadIdx.x;
  int wave = tid >> 5;
  int rw = wave >> 1;             // 0..3 -> row group of 32
  int cw = wave & 1;              // 0..1 -> col group of 64
  int m0 = blockIdx.y * 128;
  int v0 = blockIdx.x * 128;

  const bf16* Ag = Aouts + (size_t)m0 * H_;   // 128 rows x 512
  const bf16* Bg = Wb    + (size_t)v0 * H_;   // 128 rows (=cols of logits) x 512

  // copy chunk assignment: 128 rows x 4 sixteen-byte chunks = 512 chunks; 2/thread
  int ch0 = tid * 2;
  int row0 = ch0 >> 2, c40 = (ch0 & 3) * 8;
  int row1 = (ch0 + 1) >> 2, c41 = ((ch0 + 1) & 3) * 8;

  // prologue: stage 0 into buffer 0
  {
    const bf16* Ak = Ag; const bf16* Bk = Bg;
    async_copy16(Ak + row0 * H_ + c40, &Asl[0][row0 * LDST + c40]);
    async_copy16(Bk + row0 * H_ + c40, &Bsl[0][row0 * LDST + c40]);
    async_copy16(Ak + row1 * H_ + c41, &Asl[0][row1 * LDST + c41]);
    async_copy16(Bk + row1 * H_ + c41, &Bsl[0][row1 * LDST + c41]);
  }

  v8f zero = {};
  v8f acc[2][4];
#pragma unroll
  for (int a = 0; a < 2; a++)
#pragma unroll
    for (int b = 0; b < 4; b++) acc[a][b] = zero;

  for (int ks = 0; ks < 16; ks++) {
    int buf = ks & 1;
    if (ks + 1 < 16) {           // issue next stage into other buffer
      int nb = buf ^ 1;
      const bf16* Ak = Ag + (ks + 1) * 32;
      const bf16* Bk = Bg + (ks + 1) * 32;
      async_copy16(Ak + row0 * H_ + c40, &Asl[nb][row0 * LDST + c40]);
      async_copy16(Bk + row0 * H_ + c40, &Bsl[nb][row0 * LDST + c40]);
      async_copy16(Ak + row1 * H_ + c41, &Asl[nb][row1 * LDST + c41]);
      async_copy16(Bk + row1 * H_ + c41, &Bsl[nb][row1 * LDST + c41]);
      async_wait_le4();          // our stage-ks copies done (4 newer in flight)
    } else {
      async_wait_le0();
    }
    __syncthreads();             // everyone's stage-ks data visible in LDS

    const bf16* Ab = &Asl[buf][0];
    const bf16* Bb = &Bsl[buf][0];
    v16bf Af0 = load_fragA(Ab + (rw * 32 +  0) * LDST, LDST);
    v16bf Af1 = load_fragA(Ab + (rw * 32 + 16) * LDST, LDST);
#pragma unroll
    for (int fn = 0; fn < 4; fn++) {
      v16bf Bf = load_fragB(Bb + (cw * 64 + fn * 16) * LDST, LDST);
      acc[0][fn] = wmma_bf16(Af0, Bf, acc[0][fn]);
      acc[1][fn] = wmma_bf16(Af1, Bf, acc[1][fn]);
    }
    __syncthreads();             // done reading buf before it is re-filled
  }

  int lane = threadIdx.x & 31;
  int col = lane & 15, hf = lane >> 4;
#pragma unroll
  for (int fm = 0; fm < 2; fm++) {
#pragma unroll
    for (int fn = 0; fn < 4; fn++) {
      int v = v0 + cw * 64 + fn * 16 + col;
      float bv = bout[v];
      float* p = out + (size_t)(m0 + rw * 32 + fm * 16) * V_ + v;
#pragma unroll
      for (int i = 0; i < 8; i++) p[(size_t)(i + 8 * hf) * V_] = acc[fm][fn][i] + bv;
    }
  }
}

// ---------------- launch ----------------

extern "C" void kernel_launch(void* const* d_in, const int* in_sizes, int n_in,
                              void* d_out, int out_size, void* d_ws, size_t ws_size,
                              hipStream_t stream) {
  const float*         h0       = (const float*)d_in[0];
  const float*         c0       = (const float*)d_in[1];
  const float*         features = (const float*)d_in[2];
  const unsigned char* tmask    = (const unsigned char*)d_in[3];
  const int*           ids      = (const int*)d_in[4];
  const unsigned char* lmask    = (const unsigned char*)d_in[5];
  const float*         emb      = (const float*)d_in[6];
  const float*         Wih      = (const float*)d_in[7];
  const float*         Whh      = (const float*)d_in[8];
  const float*         bih      = (const float*)d_in[9];
  const float*         bhh      = (const float*)d_in[10];
  const float*         Wout     = (const float*)d_in[11];
  const float*         bout     = (const float*)d_in[12];
  float* logits = (float*)d_out;

  char* ws = (char*)d_ws;
  size_t off = 0;
  auto take = [&](size_t bytes) -> char* {
    char* p = ws + off;
    off += (bytes + 255) & ~(size_t)255;
    return p;
  };
  bf16* Xbf    = (bf16*)take((size_t)N_ * M1 * E_ * sizeof(bf16));
  bf16* WihB   = (bf16*)take((size_t)N_ * G4H * E_ * sizeof(bf16));
  bf16* WhhB   = (bf16*)take((size_t)N_ * G4H * H_ * sizeof(bf16));
  bf16* WoutB  = (bf16*)take((size_t)V_ * H_ * sizeof(bf16));
  float* bb    = (float*)take((size_t)N_ * G4H * sizeof(float));
  float* xg    = (float*)take((size_t)N_ * M1 * G4H * sizeof(float));
  bf16* outsB  = (bf16*)take((size_t)B_ * N_ * L_ * H_ * sizeof(bf16));
  int*  lens   = (int*)take((size_t)B_ * N_ * sizeof(int));
  (void)ws_size; (void)in_sizes; (void)n_in; (void)out_size;

  // 1) precision conversion + gather + biases/lengths
  k_f32_to_bf16<<<4096, 256, 0, stream>>>(Wih,  WihB,  N_ * G4H * E_);
  k_f32_to_bf16<<<4096, 256, 0, stream>>>(Whh,  WhhB,  N_ * G4H * H_);
  k_f32_to_bf16<<<8192, 256, 0, stream>>>(Wout, WoutB, V_ * H_);
  k_gather<<<(N_ * M1 * E_) / 256, 256, 0, stream>>>(features, emb, ids, Xbf);
  k_prep<<<(N_ * G4H) / 256, 256, 0, stream>>>(bih, bhh, bb, lmask, lens);

  // 2) input projection: xg = X @ Wih^T + (bih+bhh)
  k_gemm1<<<(N_ * 12 * 128) / 8, 256, 0, stream>>>(Xbf, WihB, bb, xg);

  // 3) recurrence (12 independent LSTMs, one WGP-block each)
  k_lstm<<<N_, 1024, 0, stream>>>(h0, c0, WhhB, xg, lens, tmask, outsB);

  // 4) vocab projection: logits = outs @ Wout^T + bout (dominant, 128x128 tiles,
  //    async-LDS double-buffered K pipeline)
  k_gemm2<<<dim3(V_ / 128, (B_ * N_ * L_) / 128), 256, 0, stream>>>(outsB, WoutB, bout, logits);
}
